// RQVAE_62586263437359
// MI455X (gfx1250) — matrix operand order repaired and checked by reading, not compiled, not run on hardware
//
#include <hip/hip_runtime.h>
#include <stdint.h>

// ---------------- problem constants ----------------
#define BATCH    131072
#define IN_DIM   768
#define LATENT   32
#define NLEVELS  3
#define CBK      256          // codebook entries per level
#define MT       32           // batch rows per workgroup
#define NTHREADS 256          // 8 wave32 waves

// ---------------- WMMA types (gfx1250, wave32) ----------------
typedef __attribute__((ext_vector_type(16))) __bf16    v16bf;
typedef __attribute__((ext_vector_type(8)))  float     v8f;
typedef __attribute__((ext_vector_type(8)))  uint32_t  v8u;

union Frag16 { v16bf v; uint16_t h[16]; uint32_t w[8]; v8u v8; };

// fp32 -> bf16 round-to-nearest-even (pure integer, host-parse safe)
__device__ __forceinline__ uint16_t f2bf(float f) {
  uint32_t u = __builtin_bit_cast(uint32_t, f);
  u += 0x7FFFu + ((u >> 16) & 1u);
  return (uint16_t)(u >> 16);
}

// A fragment: 16x32 bf16 tile from LDS (row-major [M][K]).
// Layout used here (contiguous k): lanes 0-15 = M0-15 half0, lanes 16-31 = M0-15 half1.
//   V0..V7 -> K = half*16 + 0..15
// (pair indices are contiguous -> compiler emits 2x ds_load_b128)
__device__ __forceinline__ v16bf load_a_frag(const uint16_t* __restrict__ hin,
                                             int K, int m, int half, int kk) {
  Frag16 f;
  const uint32_t* row = (const uint32_t*)(hin + m * K + kk);
  #pragma unroll
  for (int p = 0; p < 8; ++p) {
    const int pi = half * 8 + p;
    f.w[p] = row[pi];
  }
  return f.v;
}

// ---------------------------------------------------------------------------
// Weight pre-pack: fp32 W[K][N] -> bf16 in exact WMMA B-fragment stream order:
//   dst[((nt*KT + kkt)*32 + lane)*16 + e] =
//       bf16( W[kkt*32 + (lane>>4)*16 + e][nt*16 + (lane&15)] )
// ---------------------------------------------------------------------------
extern "C" __global__ void rqvae_pack_w(const float* __restrict__ W,
                                        uint16_t* __restrict__ dst,
                                        int K, int N) {
  const int idx = blockIdx.x * blockDim.x + threadIdx.x;
  if (idx >= K * N) return;
  const int KT   = K >> 5;
  const int e    = idx & 15;
  const int lane = (idx >> 4) & 31;
  const int tile = idx >> 9;
  const int nt   = tile / KT;
  const int kkt  = tile - nt * KT;
  const int n    = (nt << 4) + (lane & 15);
  const int k    = (kkt << 5) + ((lane >> 4) << 4) + e;
  dst[idx] = f2bf(W[(size_t)k * N + n]);
}

// Codebook pre-pack: emb[L][256][32] -> per-level bf16 B-layout of emb^T
// (K=32 dims, N=256 entries, KT=1) + per-entry squared norms e2.
extern "C" __global__ void rqvae_pack_cb(const float* __restrict__ cbk,
                                         uint16_t* __restrict__ dstp,  // [3][8192]
                                         float* __restrict__ e2) {     // [3][256]
  const int idx = blockIdx.x * blockDim.x + threadIdx.x;
  if (idx < NLEVELS * CBK * LATENT) {
    const int lvl  = idx >> 13;
    const int r    = idx & 8191;
    const int e    = r & 15;
    const int lane = (r >> 4) & 31;
    const int nt   = r >> 9;                      // KT = 1
    const int n    = (nt << 4) + (lane & 15);
    const int k    = ((lane >> 4) << 4) + e;
    dstp[idx] = f2bf(cbk[((size_t)lvl * CBK + n) * LATENT + k]);
  }
  if (idx < NLEVELS * CBK) {
    const float* row = cbk + (size_t)idx * LATENT;
    float s = 0.0f;
    #pragma unroll
    for (int j = 0; j < LATENT; ++j) s = fmaf(row[j], row[j], s);
    e2[idx] = s;
  }
}

// ---------------------------------------------------------------------------
// One dense layer: hout[MT][N] = act(hin[MT][K] @ W + b), Wp pre-packed bf16.
// 8 waves: each wave owns n-tiles strided by 8 and computes BOTH 16-row
// m-tiles against a single B fragment (2 WMMAs per B load -> 2x B reuse).
// ---------------------------------------------------------------------------
template <int K, int N, bool RELU, bool F32OUT>
__device__ __forceinline__ void gemm_lds(const uint16_t* __restrict__ hin,
                                         const uint16_t* __restrict__ Wp,
                                         const float* __restrict__ bias,
                                         uint16_t* __restrict__ hout,
                                         float* __restrict__ houtf,
                                         int wave, int lane) {
  constexpr int KT = K >> 5;
  constexpr int NT = N >> 4;
  __builtin_prefetch(Wp + ((wave * 32 + lane) << 4), 0, 1); // global_prefetch_b8
  const int half = lane >> 4;
  const int lm   = lane & 15;
  for (int nt = wave; nt < NT; nt += 8) {
    v8f c0 = {}, c1 = {};
    const uint16_t* wp = Wp + ((size_t)(nt * KT) * 32 + lane) * 16;
    #pragma unroll 4
    for (int kkt = 0; kkt < KT; ++kkt) {
      Frag16 fb;
      fb.v8 = *(const v8u*)(wp + (size_t)kkt * 512);         // 2x global_load_b128
      v16bf a0 = load_a_frag(hin, K, lm,      half, kkt << 5);
      v16bf a1 = load_a_frag(hin, K, 16 + lm, half, kkt << 5);
      c0 = __builtin_amdgcn_wmma_f32_16x16x32_bf16(false, a0, false, fb.v, (short)0, c0, false, false);
      c1 = __builtin_amdgcn_wmma_f32_16x16x32_bf16(false, a1, false, fb.v, (short)0, c1, false, false);
    }
    const int n  = (nt << 4) + lm;
    const float bn = bias[n];
    #pragma unroll
    for (int r = 0; r < 8; ++r) {
      const int m0 = half * 8 + r;            // C/D: VGPR r -> M = half*8 + r
      float v0 = c0[r] + bn;
      float v1 = c1[r] + bn;
      if (RELU) { v0 = fmaxf(v0, 0.0f); v1 = fmaxf(v1, 0.0f); }
      if (F32OUT) {
        houtf[m0 * N + n]        = v0;
        houtf[(m0 + 16) * N + n] = v1;
      } else {
        hout[m0 * N + n]         = f2bf(v0);
        hout[(m0 + 16) * N + n]  = f2bf(v1);
      }
    }
  }
}

// Final decoder layer (512 -> 768): write x_hat to HBM + accumulate recon MSE.
__device__ __forceinline__ float gemm_final(const uint16_t* __restrict__ hin,
                                            const uint16_t* __restrict__ Wp,
                                            const float* __restrict__ bias,
                                            const float* __restrict__ x,
                                            float* __restrict__ xhat,
                                            int rowbase, int wave, int lane) {
  constexpr int K = 512, N = IN_DIM;
  constexpr int KT = K >> 5;
  constexpr int NT = N >> 4;
  __builtin_prefetch(Wp + ((wave * 32 + lane) << 4), 0, 1);
  float acc = 0.0f;
  const int half = lane >> 4;
  const int lm   = lane & 15;
  for (int nt = wave; nt < NT; nt += 8) {
    v8f c0 = {}, c1 = {};
    const uint16_t* wp = Wp + ((size_t)(nt * KT) * 32 + lane) * 16;
    #pragma unroll 4
    for (int kkt = 0; kkt < KT; ++kkt) {
      Frag16 fb;
      fb.v8 = *(const v8u*)(wp + (size_t)kkt * 512);
      v16bf a0 = load_a_frag(hin, K, lm,      half, kkt << 5);
      v16bf a1 = load_a_frag(hin, K, 16 + lm, half, kkt << 5);
      c0 = __builtin_amdgcn_wmma_f32_16x16x32_bf16(false, a0, false, fb.v, (short)0, c0, false, false);
      c1 = __builtin_amdgcn_wmma_f32_16x16x32_bf16(false, a1, false, fb.v, (short)0, c1, false, false);
    }
    const int n  = (nt << 4) + lm;
    const float bn = bias[n];
    #pragma unroll
    for (int r = 0; r < 8; ++r) {
      const int m0 = half * 8 + r;
      const float v0 = c0[r] + bn;
      const float v1 = c1[r] + bn;
      const size_t g0 = (size_t)(rowbase + m0) * N + n;
      const size_t g1 = (size_t)(rowbase + m0 + 16) * N + n;
      xhat[g0] = v0;
      xhat[g1] = v1;
      const float d0 = v0 - x[g0];
      const float d1 = v1 - x[g1];
      acc = fmaf(d0, d0, acc);
      acc = fmaf(d1, d1, acc);
    }
  }
  return acc;
}

// ---------------- LDS arena (~109 KB of the 320 KB WGP LDS) ----------------
#define ACT0_OFF    0           // 32x768 bf16 = 48 KB  (ping)
#define ACT1_OFF    49152       // 32x768 bf16 = 48 KB  (pong)
#define SCORES_OFF  ACT1_OFF    // 32x256 fp32 = 32 KB, overlays ACT1 during VQ
#define ZRES_OFF    98304       // 32x32 fp32 residual
#define QSUM_OFF    102400      // 32x32 fp32 quantized sum
#define RDIST_OFF   106496      // 256 fp32 (argmin + loss reductions)
#define RIDX_OFF    107520      // 256 i32
#define BIDX_OFF    108544      // 32 i32
#define ZBF_OFF     108672      // 32x32 bf16 residual mirror (WMMA A operand)
#define E2L_OFF     110720      // 256 fp32 codebook norms for current level
#define ARENA_BYTES 111744

extern "C" __global__ __launch_bounds__(NTHREADS)
void rqvae_fused(const float* __restrict__ x,
                 const uint16_t* pe0, const float* eb0,
                 const uint16_t* pe1, const float* eb1,
                 const uint16_t* pe2, const float* eb2,
                 const uint16_t* pe3, const float* eb3,
                 const uint16_t* pd0, const float* db0,
                 const uint16_t* pd1, const float* db1,
                 const uint16_t* pd2, const float* db2,
                 const uint16_t* pd3, const float* db3,
                 const float* __restrict__ codebooks,
                 const uint16_t* __restrict__ pcb,   // packed codebook^T, [3][8192]
                 const float* __restrict__ e2g,      // codebook norms, [3][256]
                 float* __restrict__ xhat, float* __restrict__ codes,
                 float* __restrict__ ws) {
  __shared__ __align__(16) char arena[ARENA_BYTES];
  uint16_t* act0   = (uint16_t*)(arena + ACT0_OFF);
  uint16_t* act1   = (uint16_t*)(arena + ACT1_OFF);
  float*    scores = (float*)(arena + SCORES_OFF);
  float*    zres   = (float*)(arena + ZRES_OFF);
  float*    qsum   = (float*)(arena + QSUM_OFF);
  float*    rdst   = (float*)(arena + RDIST_OFF);
  int*      ridx   = (int*)(arena + RIDX_OFF);
  int*      bidx   = (int*)(arena + BIDX_OFF);
  uint16_t* zbf    = (uint16_t*)(arena + ZBF_OFF);
  float*    e2l    = (float*)(arena + E2L_OFF);

  const int tid  = threadIdx.x;
  const int wave = tid >> 5;
  const int lane = tid & 31;
  const int rowbase = blockIdx.x * MT;

  // ---- stage input tile x[rowbase:rowbase+32, :] as bf16 into LDS ----
  {
    const size_t base = (size_t)rowbase * IN_DIM;
    for (int i = tid; i < MT * IN_DIM; i += NTHREADS) act0[i] = f2bf(x[base + i]);
  }
  __syncthreads();

  // ---- encoder: 768 -> 512 -> 256 -> 128 -> 32 (last layer no relu, fp32 out) ----
  gemm_lds<768, 512, true,  false>(act0, pe0, eb0, act1, nullptr, wave, lane); __syncthreads();
  gemm_lds<512, 256, true,  false>(act1, pe1, eb1, act0, nullptr, wave, lane); __syncthreads();
  gemm_lds<256, 128, true,  false>(act0, pe2, eb2, act1, nullptr, wave, lane); __syncthreads();
  gemm_lds<128,  32, false, true >(act1, pe3, eb3, nullptr, zres, wave, lane); __syncthreads();

  // ---- residual VQ over 3 levels, scores via WMMA ----
  for (int i = tid; i < MT * LATENT; i += NTHREADS) {
    qsum[i] = 0.0f;
    zbf[i]  = f2bf(zres[i]);       // bf16 residual mirror (A operand)
  }
  __syncthreads();

  const int half = lane >> 4;
  const int lm   = lane & 15;
  for (int lvl = 0; lvl < NLEVELS; ++lvl) {
    // stage this level's codebook norms
    for (int i = tid; i < CBK; i += NTHREADS) e2l[i] = e2g[lvl * CBK + i];
    __syncthreads();

    // scores[32][256] = e2 - 2 * (res @ emb^T); x^2 term dropped (row-constant)
    {
      const uint16_t* pcbl = pcb + (size_t)lvl * (CBK * LATENT);
      for (int nt = wave; nt < (CBK >> 4); nt += 8) {
        Frag16 fb;
        fb.v8 = *(const v8u*)(pcbl + ((size_t)nt * 32 + lane) * 16);
        v16bf a0 = load_a_frag(zbf, LATENT, lm,      half, 0);
        v16bf a1 = load_a_frag(zbf, LATENT, 16 + lm, half, 0);
        v8f c0 = {}, c1 = {};
        c0 = __builtin_amdgcn_wmma_f32_16x16x32_bf16(false, a0, false, fb.v, (short)0, c0, false, false);
        c1 = __builtin_amdgcn_wmma_f32_16x16x32_bf16(false, a1, false, fb.v, (short)0, c1, false, false);
        const int n  = (nt << 4) + lm;
        const float en = e2l[n];
        #pragma unroll
        for (int r = 0; r < 8; ++r) {
          const int m0 = half * 8 + r;
          scores[m0 * CBK + n]        = fmaf(-2.0f, c0[r], en);
          scores[(m0 + 16) * CBK + n] = fmaf(-2.0f, c1[r], en);
        }
      }
    }
    __syncthreads();

    // argmin scan: thread (row, chunk-of-32-entries)
    const int row   = tid & 31;
    const int chunk = tid >> 5;
    {
      const float* srow = scores + row * CBK + chunk * 32;
      float best = 3.402823466e38f;
      int   bi   = chunk * 32;
      for (int e = 0; e < 32; ++e) {
        const float d = srow[e];
        if (d < best) { best = d; bi = chunk * 32 + e; }  // first-min, ascending e
      }
      rdst[tid] = best; ridx[tid] = bi;
    }
    __syncthreads();

    if (chunk == 0) { // one thread per row reduces 8 candidates, tie -> lowest idx
      float bb = rdst[row]; int bbi = ridx[row];
      for (int c2 = 1; c2 < 8; ++c2) {
        const float d2 = rdst[c2 * 32 + row];
        const int   i2 = ridx[c2 * 32 + row];
        if (d2 < bb || (d2 == bb && i2 < bbi)) { bb = d2; bbi = i2; }
      }
      bidx[row] = bbi;
      codes[(size_t)(rowbase + row) * NLEVELS + lvl] = (float)bbi;
    }
    __syncthreads();

    // residual / qsum / bf16-mirror update; selected entries read from L2
    {
      const float* cbg = codebooks + (size_t)lvl * CBK * LATENT;
      for (int i = tid; i < MT * LATENT; i += NTHREADS) {
        const int rr = i >> 5, jj = i & 31;
        const float q  = cbg[bidx[rr] * LATENT + jj];
        const float nr = zres[i] - q;
        zres[i] = nr;
        qsum[i] += q;
        zbf[i]  = f2bf(nr);
      }
    }
    __syncthreads();
  }

  // VQ loss partial: mean((z - q)^2) -> final residual squared
  float vq_local = 0.0f;
  for (int i = tid; i < MT * LATENT; i += NTHREADS) {
    const float r = zres[i];
    vq_local = fmaf(r, r, vq_local);
  }

  // decoder input: bf16(qsum) into act1 (scores region is dead now)
  for (int i = tid; i < MT * LATENT; i += NTHREADS) act1[i] = f2bf(qsum[i]);
  __syncthreads();

  // ---- decoder: 32 -> 128 -> 256 -> 512 -> 768 (last layer direct to HBM) ----
  gemm_lds< 32, 128, true, false>(act1, pd0, db0, act0, nullptr, wave, lane); __syncthreads();
  gemm_lds<128, 256, true, false>(act0, pd1, db1, act1, nullptr, wave, lane); __syncthreads();
  gemm_lds<256, 512, true, false>(act1, pd2, db2, act0, nullptr, wave, lane); __syncthreads();
  float recon_local = gemm_final(act0, pd3, db3, x, xhat, rowbase, wave, lane);
  __syncthreads();

  // ---- block reduction of the two loss partials ----
  rdst[tid] = recon_local;
  __syncthreads();
  for (int s = 128; s > 0; s >>= 1) { if (tid < s) rdst[tid] += rdst[tid + s]; __syncthreads(); }
  if (tid == 0) atomicAdd(&ws[0], rdst[0]);
  __syncthreads();
  rdst[tid] = vq_local;
  __syncthreads();
  for (int s = 128; s > 0; s >>= 1) { if (tid < s) rdst[tid] += rdst[tid + s]; __syncthreads(); }
  if (tid == 0) atomicAdd(&ws[1], rdst[0]);
}

extern "C" __global__ void rqvae_zero(float* ws) {
  if (threadIdx.x < 2) ws[threadIdx.x] = 0.0f;
}

extern "C" __global__ void rqvae_finalize(const float* __restrict__ ws,
                                          float* __restrict__ loss,
                                          float* __restrict__ recon) {
  const float r = ws[0] / 100663296.0f;  // B*768 (exact in fp32)
  const float v = ws[1] / 4194304.0f;    // B*32
  recon[0] = r;
  loss[0]  = r + 0.5f * v;               // recon + BETA*(commit + code), commit==code fwd
}

extern "C" void kernel_launch(void* const* d_in, const int* in_sizes, int n_in,
                              void* d_out, int out_size, void* d_ws, size_t ws_size,
                              hipStream_t stream) {
  (void)in_sizes; (void)n_in; (void)out_size; (void)ws_size;
  // setup_inputs() dict order: x, (ew0,eb0,..,ew3,eb3), (dw0,db0,..,dw3,db3), codebooks
  const float* x   = (const float*)d_in[0];
  const float* ew0 = (const float*)d_in[1];
  const float* eb0 = (const float*)d_in[2];
  const float* ew1 = (const float*)d_in[3];
  const float* eb1 = (const float*)d_in[4];
  const float* ew2 = (const float*)d_in[5];
  const float* eb2 = (const float*)d_in[6];
  const float* ew3 = (const float*)d_in[7];
  const float* eb3 = (const float*)d_in[8];
  const float* dw0 = (const float*)d_in[9];
  const float* db0 = (const float*)d_in[10];
  const float* dw1 = (const float*)d_in[11];
  const float* db1 = (const float*)d_in[12];
  const float* dw2 = (const float*)d_in[13];
  const float* db2 = (const float*)d_in[14];
  const float* dw3 = (const float*)d_in[15];
  const float* db3 = (const float*)d_in[16];
  const float* cbk = (const float*)d_in[17];

  float* out    = (float*)d_out;
  float* xhat   = out;                              // [B,768]
  float* lossp  = out + (size_t)BATCH * IN_DIM;     // scalar
  float* reconp = lossp + 1;                        // scalar
  float* codesp = reconp + 1;                       // [B,3]
  float* ws     = (float*)d_ws;                     // [0..1]: loss accumulators

  // ws layout: 64B accumulators | packed bf16 weights | packed codebooks | e2
  uint16_t* wb  = (uint16_t*)((char*)d_ws + 64);
  uint16_t* pe0 = wb + 0;        // 768*512
  uint16_t* pe1 = wb + 393216;   // 512*256
  uint16_t* pe2 = wb + 524288;   // 256*128
  uint16_t* pe3 = wb + 557056;   // 128*32
  uint16_t* pd0 = wb + 561152;   //  32*128
  uint16_t* pd1 = wb + 565248;   // 128*256
  uint16_t* pd2 = wb + 598016;   // 256*512
  uint16_t* pd3 = wb + 729088;   // 512*768  (weights end: 1122304 elems)
  uint16_t* pcb = (uint16_t*)((char*)d_ws + 2244672);  // 3*8192 bf16 = 48 KB
  float*    e2g = (float*)((char*)d_ws + 2293824);     // 3*256 fp32 = 3 KB

  rqvae_zero<<<1, 32, 0, stream>>>(ws);
  rqvae_pack_w<<<(768 * 512 + 255) / 256, 256, 0, stream>>>(ew0, pe0, 768, 512);
  rqvae_pack_w<<<(512 * 256 + 255) / 256, 256, 0, stream>>>(ew1, pe1, 512, 256);
  rqvae_pack_w<<<(256 * 128 + 255) / 256, 256, 0, stream>>>(ew2, pe2, 256, 128);
  rqvae_pack_w<<<(128 *  32 + 255) / 256, 256, 0, stream>>>(ew3, pe3, 128, 32);
  rqvae_pack_w<<<( 32 * 128 + 255) / 256, 256, 0, stream>>>(dw0, pd0, 32, 128);
  rqvae_pack_w<<<(128 * 256 + 255) / 256, 256, 0, stream>>>(dw1, pd1, 128, 256);
  rqvae_pack_w<<<(256 * 512 + 255) / 256, 256, 0, stream>>>(dw2, pd2, 256, 512);
  rqvae_pack_w<<<(512 * 768 + 255) / 256, 256, 0, stream>>>(dw3, pd3, 512, 768);
  rqvae_pack_cb<<<(NLEVELS * CBK * LATENT + 255) / 256, 256, 0, stream>>>(cbk, pcb, e2g);

  rqvae_fused<<<dim3(BATCH / MT), dim3(NTHREADS), 0, stream>>>(
      x, pe0, eb0, pe1, eb1, pe2, eb2, pe3, eb3,
      pd0, db0, pd1, db1, pd2, db2, pd3, db3,
      cbk, pcb, e2g, xhat, codesp, ws);
  rqvae_finalize<<<1, 1, 0, stream>>>(ws, lossp, reconp);
}
